// SimpleRNN_75376676044943
// MI455X (gfx1250) — matrix operand order repaired and checked by reading, not compiled, not run on hardware
//
#include <hip/hip_runtime.h>
#include <hip/hip_bf16.h>

// ---------------------------------------------------------------------------
// Types for CDNA5 WMMA (wave32): D(16x16 f32) = A(16x32 bf16) x B(32x16 bf16) + C
// ---------------------------------------------------------------------------
typedef __attribute__((ext_vector_type(16))) __bf16 v16bf;
typedef __attribute__((ext_vector_type(8)))  float  v8f;

#define RNN_B 32
#define RNN_T 512
#define RNN_I 512
#define RNN_H 1024
#define RNN_O 512
#define RNN_BLOCKS 8    // persistent blocks for recurrence: 8 blocks x 8 waves = 64 waves

#define WMMA_BF16(a, b, c) \
    __builtin_amdgcn_wmma_f32_16x16x32_bf16(false, (a), false, (b), (short)0, (c), false, false)

// float -> bf16 (round-to-nearest-even), raw u16 storage
__device__ __forceinline__ unsigned short f2bf(float f) {
    unsigned u = __float_as_uint(f);
    unsigned r = u + 0x7FFFu + ((u >> 16) & 1u);
    return (unsigned short)(r >> 16);
}

// A-matrix 16x32 bf16 tile load (row0 = first M row, kt = first K).
// ISA layout: lanes 0-15 row M=lane hold K {kt+0..7, kt+16..23},
//             lanes 16-31 row M=lane-16 hold K {kt+8..15, kt+24..31}.
__device__ __forceinline__ v16bf load_a_tile(const unsigned short* base, int stride,
                                             int row0, int kt) {
    int lane  = threadIdx.x & 31;
    int r     = row0 + (lane & 15);
    int half8 = (lane >> 4) << 3; // 0 or 8
    const unsigned short* p = base + (size_t)r * stride + kt + half8;
    union { v16bf v; uint4 u[2]; } t;
    t.u[0] = *(const uint4*)(p);
    t.u[1] = *(const uint4*)(p + 16);
    return t.v;
}

// B-matrix 32x16 bf16 tile load from W stored row-major [N][K] (row n = output col).
// ISA layout: lanes 0-15 col N=lane hold K {kt+0..7, kt+16..23}; lanes 16-31 hold K {kt+8..15, kt+24..31}.
__device__ __forceinline__ v16bf load_b_tile(const unsigned short* base, int stride,
                                             int col0, int kt) {
    int lane   = threadIdx.x & 31;
    int c      = col0 + (lane & 15);
    int half16 = (lane >> 4) << 3; // 0 or 8
    const unsigned short* p = base + (size_t)c * stride + kt + half16;
    union { v16bf v; uint4 u[2]; } t;
    t.u[0] = *(const uint4*)(p);
    t.u[1] = *(const uint4*)(p + 16);
    return t.v;
}

// ---------------------------------------------------------------------------
// f32 -> bf16 bulk convert (grid-stride)
// ---------------------------------------------------------------------------
__global__ void f32_to_bf16_kernel(const float* __restrict__ src,
                                   unsigned short* __restrict__ dst, int n) {
    int i      = blockIdx.x * blockDim.x + threadIdx.x;
    int stride = gridDim.x * blockDim.x;
    for (; i < n; i += stride) dst[i] = f2bf(src[i]);
}

// ---------------------------------------------------------------------------
// Init: fused bias (b_ih + b_hh), zero h0 ping buffer, reset grid-barrier counter
// ---------------------------------------------------------------------------
__global__ void rnn_init_kernel(const float* __restrict__ b_ih,
                                const float* __restrict__ b_hh,
                                float* __restrict__ bias_sum,
                                unsigned short* __restrict__ hb0,
                                unsigned* __restrict__ cnt) {
    int tid = threadIdx.x; // one block of 1024
    bias_sum[tid] = b_ih[tid] + b_hh[tid];
    for (int i = tid; i < RNN_B * RNN_H; i += 1024) hb0[i] = 0;
    if (tid == 0) *cnt = 0u;
}

// ---------------------------------------------------------------------------
// Generic bf16 WMMA GEMM with f32 bias, 2x2 register blocking:
//   out[M][N] (f32) = A[M][K] (bf16) * Bw[N][K]^T (bf16) + bias[N]
// Each wave owns a 32x32 output patch (4 accumulator tiles, 4 independent
// WMMA chains); 8 b128 loads feed 4 WMMAs per K-step. 8 waves per block.
// ---------------------------------------------------------------------------
__global__ void wmma_gemm_bias_kernel(const unsigned short* __restrict__ A,
                                      const unsigned short* __restrict__ Bw,
                                      const float* __restrict__ bias,
                                      float* __restrict__ out,
                                      int M, int N, int K) {
    int wave = blockIdx.x * (blockDim.x >> 5) + (threadIdx.x >> 5);
    int n32  = N >> 5;
    int mj   = wave / n32;
    int nj   = wave - mj * n32;
    if (mj >= (M >> 5)) return;
    int m0 = mj << 5, n0 = nj << 5;

    v8f acc00 = {}, acc01 = {}, acc10 = {}, acc11 = {};
    for (int kt = 0; kt < K; kt += 32) {
        v16bf a0 = load_a_tile(A, K, m0,      kt);
        v16bf a1 = load_a_tile(A, K, m0 + 16, kt);
        v16bf b0 = load_b_tile(Bw, K, n0,      kt);
        v16bf b1 = load_b_tile(Bw, K, n0 + 16, kt);
        acc00 = WMMA_BF16(a0, b0, acc00);
        acc01 = WMMA_BF16(a0, b1, acc01);
        acc10 = WMMA_BF16(a1, b0, acc10);
        acc11 = WMMA_BF16(a1, b1, acc11);
    }

    // C/D layout: lane holds column n; VGPR v holds row m0+v (lanes>=16: +8)
    int lane = threadIdx.x & 31;
    int nl   = lane & 15;
    int mb   = (lane >> 4) << 3;
    float bv0 = bias[n0 + nl];
    float bv1 = bias[n0 + 16 + nl];
#pragma unroll
    for (int v = 0; v < 8; ++v) {
        size_t r0 = (size_t)(m0 + mb + v) * N;
        size_t r1 = (size_t)(m0 + 16 + mb + v) * N;
        out[r0 + n0 + nl]      = acc00[v] + bv0;
        out[r0 + n0 + 16 + nl] = acc01[v] + bv1;
        out[r1 + n0 + nl]      = acc10[v] + bv0;
        out[r1 + n0 + 16 + nl] = acc11[v] + bv1;
    }
}

// ---------------------------------------------------------------------------
// Grid-wide barrier for the persistent recurrence kernel (monotonic counter)
// ---------------------------------------------------------------------------
__device__ __forceinline__ void grid_sync(unsigned* cnt, unsigned expected) {
    __threadfence();
    __syncthreads();
    if (threadIdx.x == 0) {
        atomicAdd(cnt, 1u);
        volatile unsigned* vc = (volatile unsigned*)cnt;
        while (*vc < expected) { __builtin_amdgcn_s_sleep(2); }
    }
    __syncthreads();
    __threadfence();
}

// ---------------------------------------------------------------------------
// Persistent Elman recurrence:  h_t = relu(xp[:,t,:] + h_{t-1} @ W_hh^T)
// 8 blocks x 8 waves = 64 waves; wave w owns hidden columns [16w, 16w+16) and
// BOTH batch tiles (m=0..15, 16..31), so each W_hh tile load feeds 2 WMMAs and
// W_hh streams from L2 exactly once per step (2 MB/step). K is split into two
// interleaved accumulator chains per M-tile -> 4 independent WMMA chains,
// halving the serialized dependency depth of the latency-critical step.
// h ping-pongs between two bf16 global buffers (L2-resident, 64 KB each).
// ---------------------------------------------------------------------------
__global__ void rnn_recur_kernel(const unsigned short* __restrict__ Whh,
                                 const float* __restrict__ xp,
                                 unsigned short* __restrict__ hs,
                                 unsigned short* __restrict__ hb0,
                                 unsigned short* __restrict__ hb1,
                                 unsigned* __restrict__ cnt) {
    int wave = blockIdx.x * (blockDim.x >> 5) + (threadIdx.x >> 5); // 0..63
    int n0   = wave << 4;                                          // 0..1008
    int lane = threadIdx.x & 31;
    int n    = n0 + (lane & 15);
    int mb   = (lane >> 4) << 3;

    for (int t = 0; t < RNN_T; ++t) {
        const unsigned short* hprev = (t & 1) ? hb1 : hb0;
        unsigned short*       hnext = (t & 1) ? hb0 : hb1;

        v8f acc0a = {}, acc0b = {}, acc1a = {}, acc1b = {};
        for (int kt = 0; kt < RNN_H; kt += 64) {
            v16bf a0 = load_a_tile(hprev, RNN_H, 0,  kt);
            v16bf a1 = load_a_tile(hprev, RNN_H, 16, kt);
            v16bf b  = load_b_tile(Whh, RNN_H, n0, kt);
            acc0a = WMMA_BF16(a0, b, acc0a);
            acc1a = WMMA_BF16(a1, b, acc1a);

            v16bf a0x = load_a_tile(hprev, RNN_H, 0,  kt + 32);
            v16bf a1x = load_a_tile(hprev, RNN_H, 16, kt + 32);
            v16bf bx  = load_b_tile(Whh, RNN_H, n0, kt + 32);
            if (kt + 64 < RNN_H)
                __builtin_prefetch(Whh + (size_t)n0 * RNN_H + kt + 64, 0, 1);
            acc0b = WMMA_BF16(a0x, bx, acc0b);
            acc1b = WMMA_BF16(a1x, bx, acc1b);
        }

#pragma unroll
        for (int v = 0; v < 8; ++v) {
            int m0v = mb + v;          // batch rows 0..15
            int m1v = 16 + mb + v;     // batch rows 16..31
            size_t g0 = ((size_t)m0v * RNN_T + t) * RNN_H + n;
            size_t g1 = ((size_t)m1v * RNN_T + t) * RNN_H + n;
            float r0 = fmaxf(acc0a[v] + acc0b[v] + xp[g0], 0.0f);
            float r1 = fmaxf(acc1a[v] + acc1b[v] + xp[g1], 0.0f);
            unsigned short h0 = f2bf(r0);
            unsigned short h1 = f2bf(r1);
            hnext[m0v * RNN_H + n] = h0;
            hnext[m1v * RNN_H + n] = h1;
            hs[g0] = h0;
            hs[g1] = h1;
        }

        grid_sync(cnt, (unsigned)(RNN_BLOCKS * (t + 1)));
    }
}

// ---------------------------------------------------------------------------
// Host-side orchestration
// ---------------------------------------------------------------------------
extern "C" void kernel_launch(void* const* d_in, const int* in_sizes, int n_in,
                              void* d_out, int out_size, void* d_ws, size_t ws_size,
                              hipStream_t stream) {
    (void)in_sizes; (void)n_in; (void)out_size; (void)ws_size;

    const float* x     = (const float*)d_in[0]; // [B,T,I]
    const float* W_ih  = (const float*)d_in[1]; // [H,I]
    const float* W_hh  = (const float*)d_in[2]; // [H,H]
    const float* b_ih  = (const float*)d_in[3]; // [H]
    const float* b_hh  = (const float*)d_in[4]; // [H]
    const float* W_out = (const float*)d_in[5]; // [O,H]
    const float* b_out = (const float*)d_in[6]; // [O]
    float* out = (float*)d_out;                 // [B,T,O]

    const int BT = RNN_B * RNN_T; // 16384

    // Workspace carve-up (256B aligned slices)
    char* w = (char*)d_ws;
    size_t off = 0;
    auto alloc = [&](size_t bytes) -> void* {
        void* p = w + off;
        off = (off + bytes + 255) & ~(size_t)255;
        return p;
    };
    float*          xp       = (float*)alloc((size_t)BT * RNN_H * 4);          // 64 MB
    unsigned short* x_bf     = (unsigned short*)alloc((size_t)BT * RNN_I * 2); // 16 MB
    unsigned short* Wih_bf   = (unsigned short*)alloc((size_t)RNN_H * RNN_I * 2);
    unsigned short* Whh_bf   = (unsigned short*)alloc((size_t)RNN_H * RNN_H * 2);
    unsigned short* Wout_bf  = (unsigned short*)alloc((size_t)RNN_O * RNN_H * 2);
    unsigned short* hs_bf    = (unsigned short*)alloc((size_t)BT * RNN_H * 2); // 32 MB
    unsigned short* hb0      = (unsigned short*)alloc((size_t)RNN_B * RNN_H * 2);
    unsigned short* hb1      = (unsigned short*)alloc((size_t)RNN_B * RNN_H * 2);
    float*          bias_sum = (float*)alloc((size_t)RNN_H * 4);
    unsigned*       cnt      = (unsigned*)alloc(256);

    // 1) bf16 staging of x and all weight matrices
    f32_to_bf16_kernel<<<2048, 256, 0, stream>>>(x,     x_bf,    BT * RNN_I);
    f32_to_bf16_kernel<<<1024, 256, 0, stream>>>(W_ih,  Wih_bf,  RNN_H * RNN_I);
    f32_to_bf16_kernel<<<1024, 256, 0, stream>>>(W_hh,  Whh_bf,  RNN_H * RNN_H);
    f32_to_bf16_kernel<<<1024, 256, 0, stream>>>(W_out, Wout_bf, RNN_O * RNN_H);

    // 2) fused bias, zero h0, reset barrier counter
    rnn_init_kernel<<<1, 1024, 0, stream>>>(b_ih, b_hh, bias_sum, hb0, cnt);

    // 3) xp = x @ W_ih^T + (b_ih + b_hh)   [16384 x 1024, K=512]
    {
        int jobs   = (BT >> 5) * (RNN_H >> 5); // 512*32 = 16384 wave-jobs
        int blocks = jobs / 8;                 // 2048 blocks of 8 waves
        wmma_gemm_bias_kernel<<<blocks, 256, 0, stream>>>(
            x_bf, Wih_bf, bias_sum, xp, BT, RNN_H, RNN_I);
    }

    // 4) sequential recurrence (persistent, grid-synced each timestep)
    rnn_recur_kernel<<<RNN_BLOCKS, 256, 0, stream>>>(Whh_bf, xp, hs_bf, hb0, hb1, cnt);

    // 5) out = hs @ W_out^T + b_out        [16384 x 512, K=1024]
    {
        int jobs   = (BT >> 5) * (RNN_O >> 5); // 512*16 = 8192 wave-jobs
        int blocks = jobs / 8;                 // 1024 blocks
        wmma_gemm_bias_kernel<<<blocks, 256, 0, stream>>>(
            hs_bf, Wout_bf, b_out, out, BT, RNN_O, RNN_H);
    }
}
